// MultiheadAttention_75488345195380
// MI455X (gfx1250) — compile-verified, matrix operand order checked by reading, not run on hardware
//
#include <hip/hip_runtime.h>

// ---------------------------------------------------------------------------
// MI455X (gfx1250) multi-head attention: bf16 WMMA + async global->LDS copies.
// B=4, S=2048, D=1024, H=16, DK=64.
// Softmax runs in the base-2 domain: the Q projection epilogue pre-scales by
// (1/sqrt(DK))*log2(e), so flash attention uses bare v_exp_f32 via
// __builtin_amdgcn_exp2f. Cross-lane reductions use single-instruction
// ds_swizzle_b32 (xor mode) instead of shfl's bpermute+address-math.
// ---------------------------------------------------------------------------

typedef __attribute__((ext_vector_type(16))) __bf16       v16bf;
typedef __attribute__((ext_vector_type(8)))  float        v8f;
typedef __attribute__((ext_vector_type(8)))  unsigned int v8u;

#define DEV static __device__ __forceinline__

constexpr int BT  = 4;
constexpr int ST  = 2048;
constexpr int DT  = 1024;
constexpr int HT  = 16;
constexpr int DKT = 64;
constexpr int NTOK = BT * ST;   // 8192 tokens

// fp32 -> bf16 bits, round-to-nearest-even
DEV unsigned bfbits(float f) {
  unsigned u = __float_as_uint(f);
  return (u + 0x7FFFu + ((u >> 16) & 1u)) >> 16;
}
DEV unsigned pack2bf(float lo, float hi) { return bfbits(lo) | (bfbits(hi) << 16); }

// ds_swizzle_b32 xor-exchange: offset15=0 group-of-32 mode,
// imm = {xor_mask[14:10], or_mask[9:5]=0, and_mask[4:0]=0x1f}
template <int XORM>
DEV float swz_xor(float x) {
  constexpr int IMM = (XORM << 10) | 0x1f;
  return __int_as_float(__builtin_amdgcn_ds_swizzle(__float_as_int(x), IMM));
}
// max/sum across each 16-lane half (xor masks 1,2,4,8 stay in-half)
DEV float redmax16(float v) {
  v = fmaxf(v, swz_xor<1>(v));
  v = fmaxf(v, swz_xor<2>(v));
  v = fmaxf(v, swz_xor<4>(v));
  v = fmaxf(v, swz_xor<8>(v));
  return v;
}
DEV float redsum16(float v) {
  v += swz_xor<1>(v);
  v += swz_xor<2>(v);
  v += swz_xor<4>(v);
  v += swz_xor<8>(v);
  return v;
}

// D = A(16x32 bf16) x B(32x16 bf16) + C(16x16 f32), wave32
DEV v8f wmma_bf16(v8u a, v8u b, v8f c) {
  return __builtin_amdgcn_wmma_f32_16x16x32_bf16(
      false, __builtin_bit_cast(v16bf, a),
      false, __builtin_bit_cast(v16bf, b),
      (short)0, c, false, false);
}

// A/B fragment from bf16 row data: lane points at (row, (lane>>4)*8);
// vgprs 0..3 <- k pairs at +0..7, vgprs 4..7 <- k pairs at +16..23 (ISA 7.12.2)
DEV v8u frag_bf16(const unsigned short* p) {
  uint4 a = *(const uint4*)p;
  uint4 b = *(const uint4*)(p + 16);
  v8u r = {a.x, a.y, a.z, a.w, b.x, b.y, b.z, b.w};
  return r;
}

// Async DMA 16 bytes global -> LDS (GVS mode: saddr base + 32-bit voffset).
// Tracked by ASYNCcnt; completion enforced with s_wait_asynccnt.
DEV void async_b128(unsigned lds_byte, unsigned gbyte, const unsigned short* base) {
  asm volatile("global_load_async_to_lds_b128 %0, %1, %2"
               :: "v"(lds_byte), "v"(gbyte), "s"(base)
               : "memory");
}
DEV void wait_async0() { asm volatile("s_wait_asynccnt 0x0" ::: "memory"); }

// Generic pointer to a __shared__ object -> LDS byte offset (flat LDS aperture
// keeps the in-allocation offset in addr[31:0], ISA 10.2).
DEV unsigned lds_off(const void* p) { return (unsigned)(size_t)p; }

// ---------------------------------------------------------------------------
// fp32 -> bf16 bulk conversion (8 elements / thread)
// ---------------------------------------------------------------------------
__global__ __launch_bounds__(256)
void f32_to_bf16_kernel(const float* __restrict__ in,
                        unsigned short* __restrict__ out, int n8) {
  int i = blockIdx.x * 256 + threadIdx.x;
  if (i < n8) {
    float4 a = ((const float4*)in)[2 * i];
    float4 b = ((const float4*)in)[2 * i + 1];
    uint4 o;
    o.x = pack2bf(a.x, a.y); o.y = pack2bf(a.z, a.w);
    o.z = pack2bf(b.x, b.y); o.w = pack2bf(b.z, b.w);
    ((uint4*)out)[i] = o;
  }
}

// ---------------------------------------------------------------------------
// bf16 GEMM + bias: Out[m,n] = (sum_k A[m,k]*W[n,k] + bias[n]) * oscale
// Block tile 128x128, K-step 32. 256 threads = 8 waves, wave tile 32x64.
// Tiles staged with async global->LDS b128, double buffered: issue k+1 after
// the barrier, compute k while it flies, wait asynccnt 0 at top of next iter.
// ---------------------------------------------------------------------------
template <bool OUT_PERM>
__global__ __launch_bounds__(256)
void gemm_bf16_kernel(const unsigned short* __restrict__ A,
                      const unsigned short* __restrict__ Wb,
                      const float* __restrict__ bias,
                      void* __restrict__ Out, float oscale) {
  constexpr int LD   = 40;         // ushorts per tile row (80 B, 16B multiple)
  constexpr int TILE = 128 * LD;   // ushorts per A or B tile
  __shared__ __align__(16) unsigned short lds[2 * 2 * TILE];   // 40 KB

  const int t = threadIdx.x, l = t & 31, w = t >> 5;
  const int h2 = l >> 4, l16 = l & 15;
  const int wm = w >> 1, wn = w & 1;
  const int bm = blockIdx.x * 128, bn = blockIdx.y * 128;
  const unsigned lbase = lds_off(lds);

  const v8f zf = {0.f, 0.f, 0.f, 0.f, 0.f, 0.f, 0.f, 0.f};
  v8f acc[2][4];
#pragma unroll
  for (int mt = 0; mt < 2; ++mt)
#pragma unroll
    for (int nt = 0; nt < 4; ++nt) acc[mt][nt] = zf;

  // issue async copies of the 128x32 bf16 A and B tiles for k0 into buf
  auto issue = [&](int buf, int k0) {
#pragma unroll
    for (int j = 0; j < 2; ++j) {
      int idx = t + j * 256;                // 512 chunks of 16B per tile
      int r = idx >> 2, c = idx & 3;
      unsigned loff = lbase + (unsigned)buf * (2 * TILE * 2) + r * (LD * 2) + c * 16;
      async_b128(loff,            (unsigned)(((bm + r) * DT + k0 + c * 8) * 2), A);
      async_b128(loff + TILE * 2, (unsigned)(((bn + r) * DT + k0 + c * 8) * 2), Wb);
    }
  };

  issue(0, 0);
  for (int kk = 0; kk < DT / 32; ++kk) {
    wait_async0();        // my async writes into buf[kk&1] have landed
    __syncthreads();      // everyone's writes landed; prev compute done
    if (kk + 1 < DT / 32) issue((kk + 1) & 1, (kk + 1) * 32);

    const unsigned short* la = lds + (kk & 1) * (2 * TILE);
    const unsigned short* lb = la + TILE;

    v8u fa[2], fb[4];
#pragma unroll
    for (int mt = 0; mt < 2; ++mt)
      fa[mt] = frag_bf16(&la[(wm * 32 + mt * 16 + l16) * LD + h2 * 8]);
#pragma unroll
    for (int nt = 0; nt < 4; ++nt)
      fb[nt] = frag_bf16(&lb[(wn * 64 + nt * 16 + l16) * LD + h2 * 8]);
#pragma unroll
    for (int mt = 0; mt < 2; ++mt)
#pragma unroll
      for (int nt = 0; nt < 4; ++nt)
        acc[mt][nt] = wmma_bf16(fa[mt], fb[nt], acc[mt][nt]);
    // ds reads complete before next iteration's barrier (consumed by wmma)
  }

  // Epilogue.  C layout: vgpr i -> M = i + 8*(lane>>4), N = lane&15.
#pragma unroll
  for (int nt = 0; nt < 4; ++nt) {
    int col = bn + wn * 64 + nt * 16 + l16;
    float bc = bias[col];
#pragma unroll
    for (int mt = 0; mt < 2; ++mt) {
#pragma unroll
      for (int i = 0; i < 8; ++i) {
        int grow = bm + wm * 32 + mt * 16 + i + 8 * h2;
        float v = (acc[mt][nt][i] + bc) * oscale;
        if constexpr (OUT_PERM) {
          int head = col >> 6, dk = col & 63;
          int bb = grow >> 11, ss = grow & 2047;
          ((unsigned short*)Out)[((size_t)(bb * HT + head) * ST + ss) * DKT + dk] =
              (unsigned short)bfbits(v);
        } else {
          ((float*)Out)[(size_t)grow * DT + col] = v;
        }
      }
    }
  }
}

// ---------------------------------------------------------------------------
// Flash attention over one (b,h): grid (S/128, B*H), 8 waves, 16 q-rows/wave,
// key tile 64. K tile staged via async DMA (overlapped with the V transpose),
// online softmax in the exp2 domain (q pre-scaled by log2e/sqrt(DK)),
// P restaged per-wave through LDS (C->A layout), O accumulated in f32.
// ---------------------------------------------------------------------------
__global__ __launch_bounds__(256)
void flash_attn_kernel(const unsigned short* __restrict__ qb,
                       const unsigned short* __restrict__ kb,
                       const unsigned short* __restrict__ vb,
                       unsigned short* __restrict__ ob) {
  constexpr int KT  = 64;
  constexpr int LDV = 72;
  constexpr int LDP = 72;
  __shared__ __align__(16) unsigned short ldsK[KT * DKT];   // [key][dk]  8 KB
  __shared__ unsigned short ldsVt[DKT * LDV];               // [dk][key]  9 KB
  __shared__ unsigned short ldsP[8 * 16 * LDP];             // per-wave P 18 KB

  const int t = threadIdx.x, l = t & 31, w = t >> 5;
  const int h2 = l >> 4, l16 = l & 15;
  const int qblk = blockIdx.x, bh = blockIdx.y;
  const int b = bh >> 4, h = bh & 15;
  const size_t base = (size_t)bh * ST * DKT;
  const unsigned base32 = (unsigned)base;          // fits: <= 16 MB elements
  const int qrow = qblk * 128 + w * 16 + l16;
  const unsigned lK = lds_off(ldsK);

  v8u aq[2];
  {
    const unsigned short* qp = qb + base + (size_t)qrow * DKT + h2 * 8;
    aq[0] = frag_bf16(qp);
    aq[1] = frag_bf16(qp + 32);
  }

  const v8f zf = {0.f, 0.f, 0.f, 0.f, 0.f, 0.f, 0.f, 0.f};
  v8f oacc[4] = {zf, zf, zf, zf};
  float mrun[8], lrun[8];
#pragma unroll
  for (int i = 0; i < 8; ++i) { mrun[i] = -3.0e38f; lrun[i] = 0.f; }

  for (int kt = 0; kt < ST / KT; ++kt) {
    const int kbase = kt * KT;
    // K tile: async DMA, 512 x 16B chunks, 2 per thread
#pragma unroll
    for (int j = 0; j < 2; ++j) {
      int idx = t + j * 256;
      int r = idx >> 3, c = idx & 7;
      async_b128(lK + r * 128 + c * 16,
                 (base32 + (unsigned)(kbase + r) * DKT + c * 8) * 2, kb);
    }
    // V tile transposed (normal loads/stores, overlapped with the DMA above)
    for (int i = t; i < KT * 32; i += 256) {
      int key = i >> 5, d2 = i & 31;
      unsigned u = *(const unsigned*)&vb[base + (size_t)(kbase + key) * DKT + d2 * 2];
      ldsVt[(d2 * 2 + 0) * LDV + key] = (unsigned short)(u & 0xffffu);
      ldsVt[(d2 * 2 + 1) * LDV + key] = (unsigned short)(u >> 16);
    }
    wait_async0();
    __syncthreads();

    // scores: already in log2 units (q pre-scaled by log2e/sqrt(DK))
    v8f sc[4];
#pragma unroll
    for (int nt = 0; nt < 4; ++nt) {
      const unsigned short* kp = &ldsK[(nt * 16 + l16) * DKT + h2 * 8];
      v8f a = zf;
      a = wmma_bf16(aq[0], frag_bf16(kp), a);
      a = wmma_bf16(aq[1], frag_bf16(kp + 32), a);
      sc[nt] = a;
    }

    // online softmax, base-2 (C-layout rows; ds_swizzle xor reductions in-half)
#pragma unroll
    for (int i = 0; i < 8; ++i) {
      float tm = fmaxf(fmaxf(sc[0][i], sc[1][i]), fmaxf(sc[2][i], sc[3][i]));
      tm = redmax16(tm);
      float mn = fmaxf(mrun[i], tm);
      float alpha = __builtin_amdgcn_exp2f(mrun[i] - mn);   // bare v_exp_f32
      float rs = 0.f;
#pragma unroll
      for (int nt = 0; nt < 4; ++nt) {
        float p = __builtin_amdgcn_exp2f(sc[nt][i] - mn);
        sc[nt][i] = p;
        rs += p;
      }
      rs = redsum16(rs);
      lrun[i] = lrun[i] * alpha + rs;
      mrun[i] = mn;
#pragma unroll
      for (int nt2 = 0; nt2 < 4; ++nt2) oacc[nt2][i] *= alpha;
    }

    // stage P (bf16) C-layout -> row-major, per-wave private LDS region
    unsigned short* pw = &ldsP[w * 16 * LDP];
#pragma unroll
    for (int nt = 0; nt < 4; ++nt)
#pragma unroll
      for (int i = 0; i < 8; ++i)
        pw[(i + 8 * h2) * LDP + nt * 16 + l16] = (unsigned short)bfbits(sc[nt][i]);
    // LDS ops from the same wave complete in order -> wave-local reuse is safe

    // O += P V  (A = P with k = key; B lane = dk via transposed V tile)
#pragma unroll
    for (int ks = 0; ks < 2; ++ks) {
      v8u ap = frag_bf16(&pw[l16 * LDP + ks * 32 + h2 * 8]);
#pragma unroll
      for (int nt2 = 0; nt2 < 4; ++nt2) {
        v8u bv = frag_bf16(&ldsVt[(nt2 * 16 + l16) * LDV + ks * 32 + h2 * 8]);
        oacc[nt2] = wmma_bf16(ap, bv, oacc[nt2]);
      }
    }
    __syncthreads();
  }

  // normalize, merge heads -> [token, D] bf16
#pragma unroll
  for (int nt2 = 0; nt2 < 4; ++nt2)
#pragma unroll
    for (int i = 0; i < 8; ++i) {
      float v = oacc[nt2][i] / lrun[i];
      int row = qblk * 128 + w * 16 + i + 8 * h2;
      int col = h * DKT + nt2 * 16 + l16;
      ob[((size_t)b * ST + row) * DT + col] = (unsigned short)bfbits(v);
    }
}

// ---------------------------------------------------------------------------
extern "C" void kernel_launch(void* const* d_in, const int* in_sizes, int n_in,
                              void* d_out, int out_size, void* d_ws, size_t ws_size,
                              hipStream_t stream) {
  (void)in_sizes; (void)n_in; (void)out_size; (void)ws_size;
  const float* Q  = (const float*)d_in[0];
  const float* K  = (const float*)d_in[1];
  const float* V  = (const float*)d_in[2];
  const float* Wq = (const float*)d_in[3];
  const float* bq = (const float*)d_in[4];
  const float* Wk = (const float*)d_in[5];
  const float* bk = (const float*)d_in[6];
  const float* Wv = (const float*)d_in[7];
  const float* bv = (const float*)d_in[8];
  const float* Wo = (const float*)d_in[9];
  const float* bo = (const float*)d_in[10];

  constexpr size_t SZX = (size_t)NTOK * DT;   // 8 M elements
  constexpr size_t SZW = (size_t)DT * DT;     // 1 M elements
  unsigned short* xb  = (unsigned short*)d_ws;   // bf16 activations (reused)
  unsigned short* wqb = xb + SZX;
  unsigned short* wkb = wqb + SZW;
  unsigned short* wvb = wkb + SZW;
  unsigned short* wob = wvb + SZW;
  unsigned short* qbb = wob + SZW;
  unsigned short* kbb = qbb + SZX;
  unsigned short* vbb = kbb + SZX;
  unsigned short* obf = xb;                      // alias: xb is dead by then

  // q pre-scale: log2(e)/sqrt(DK) so softmax runs on exp2
  const float qscale = 1.4426950408889634f * 0.125f;

  const int gw = (int)(SZW / 8 / 256);
  const int gx = (int)(SZX / 8 / 256);
  f32_to_bf16_kernel<<<gw, 256, 0, stream>>>(Wq, wqb, (int)(SZW / 8));
  f32_to_bf16_kernel<<<gw, 256, 0, stream>>>(Wk, wkb, (int)(SZW / 8));
  f32_to_bf16_kernel<<<gw, 256, 0, stream>>>(Wv, wvb, (int)(SZW / 8));
  f32_to_bf16_kernel<<<gw, 256, 0, stream>>>(Wo, wob, (int)(SZW / 8));

  dim3 ggrid(NTOK / 128, DT / 128);
  f32_to_bf16_kernel<<<gx, 256, 0, stream>>>(Q, xb, (int)(SZX / 8));
  gemm_bf16_kernel<true><<<ggrid, 256, 0, stream>>>(xb, wqb, bq, qbb, qscale);
  f32_to_bf16_kernel<<<gx, 256, 0, stream>>>(K, xb, (int)(SZX / 8));
  gemm_bf16_kernel<true><<<ggrid, 256, 0, stream>>>(xb, wkb, bk, kbb, 1.0f);
  f32_to_bf16_kernel<<<gx, 256, 0, stream>>>(V, xb, (int)(SZX / 8));
  gemm_bf16_kernel<true><<<ggrid, 256, 0, stream>>>(xb, wvb, bv, vbb, 1.0f);

  flash_attn_kernel<<<dim3(ST / 128, BT * HT), 256, 0, stream>>>(qbb, kbb, vbb, obf);
  gemm_bf16_kernel<false><<<ggrid, 256, 0, stream>>>(obf, wob, bo, d_out, 1.0f);
}
